// AttentionBlock_84859963834749
// MI455X (gfx1250) — matrix operand order repaired and checked by reading, not compiled, not run on hardware
//
#include <hip/hip_runtime.h>
#include <hip/hip_bf16.h>
#include <math.h>

// ---------------------------------------------------------------------------
// Problem constants
// ---------------------------------------------------------------------------
#define BQ 8
#define HH 256
#define WW 256
#define CC 64
#define NPIX (BQ * HH * WW)       // 524288 rows of the implicit GEMM
#define MROWS NPIX
#define MTILES (MROWS / 16)       // 32768 16-row WMMA tiles
#define PI_F 3.14159265358979f
#define EPS_F 1e-8f
#define BN_EPS_F 1e-5f

typedef __attribute__((ext_vector_type(16))) _Float16 v16h;
typedef __attribute__((ext_vector_type(8)))  float    v8f;

// stats buffer layout (in workspace)
#define S_MEAN0 0     // [64]
#define S_RSTD0 64    // [64]
#define S_MEAN1 128   // [64]
#define S_RSTD1 192   // [64]
#define S_M2 256
#define S_R2 257
#define S_M3 258
#define S_R3 259
#define S_M4 260
#define S_R4 261

// ---------------------------------------------------------------------------
// Helpers
// ---------------------------------------------------------------------------
__device__ __forceinline__ float sigmoidf_(float x) { return 1.0f / (1.0f + expf(-x)); }
__device__ __forceinline__ int wrap256(int i) { return i & 255; }  // jnp.roll wrap, 256 = 2^8

__device__ __forceinline__ v8f wmma_f16(v16h a, v16h b, v8f c) {
  // (neg_a, A, neg_b, B, c_mod, C, reuse_a, reuse_b) -> v_wmma_f32_16x16x32_f16
  return __builtin_amdgcn_wmma_f32_16x16x32_f16(false, a, false, b, (short)0, c, false, false);
}

// A fragment: 16x32 f16 tile of X[M,64] (rows row0..row0+15, ks k0..k0+31).
// ISA layout: lanes 0-15 = rows, K {0..7,16..23}; lanes 16-31 = rows, K {8..15,24..31}.
__device__ __forceinline__ v16h load_a_f32(const float* X, int row0, int k0, int lane) {
  int r  = lane & 15;
  int kh = (lane >> 4) & 1;
  const float* p  = X + (size_t)(row0 + r) * CC + k0 + kh * 8;
  const float4* p0 = (const float4*)p;          // k = kh*8 + 0..7
  const float4* p1 = (const float4*)(p + 16);   // k = 16 + kh*8 + 0..7
  float4 u0 = p0[0], u1 = p0[1], u2 = p1[0], u3 = p1[1];
  v16h a;
  a[0]  = (_Float16)u0.x; a[1]  = (_Float16)u0.y; a[2]  = (_Float16)u0.z; a[3]  = (_Float16)u0.w;
  a[4]  = (_Float16)u1.x; a[5]  = (_Float16)u1.y; a[6]  = (_Float16)u1.z; a[7]  = (_Float16)u1.w;
  a[8]  = (_Float16)u2.x; a[9]  = (_Float16)u2.y; a[10] = (_Float16)u2.z; a[11] = (_Float16)u2.w;
  a[12] = (_Float16)u3.x; a[13] = (_Float16)u3.y; a[14] = (_Float16)u3.z; a[15] = (_Float16)u3.w;
  return a;
}

// B fragment: 32x16 f16 tile of W[K=64,N=64] row-major (cols n0..n0+15, ks k0..k0+31).
// lanes 0-15 hold K {0..7,16..23}, lanes 16-31 hold K {8..15,24..31}; lane -> column.
__device__ __forceinline__ v16h load_b_w(const float* Wt, int n0, int k0, int lane) {
  int n  = n0 + (lane & 15);
  int kb = k0 + ((lane >> 4) & 1) * 8;
  v16h b;
#pragma unroll
  for (int e = 0; e < 16; ++e) b[e] = (_Float16)Wt[(size_t)(kb + e + (e >= 8 ? 8 : 0)) * CC + n];
  return b;
}

// ---------------------------------------------------------------------------
// K1: WMMA stat pass for ONE matrix product y = X @ W (bias cancels in BN).
// Launched twice (input*W0, residual*W1) to keep the live register set small:
// 8 hoisted B fragments (64 VGPRs) + 4 accumulators (32 VGPRs) — no spill.
// Deterministic per-(block,channel) partial sum / sumsq; y is never spilled.
// grid 1024 x 256 (8 waves), each wave loops tiles.
// ---------------------------------------------------------------------------
__global__ void k1_gemm_stats(const float* __restrict__ X, const float* __restrict__ Wt,
                              float* __restrict__ bp) {
  int wave = threadIdx.x >> 5, lane = threadIdx.x & 31;
  __shared__ float lds[2][8][64];  // q(sum, sumsq) x wave x channel
  for (int i = threadIdx.x; i < 2 * 8 * 64; i += 256) ((float*)lds)[i] = 0.0f;
  __syncthreads();

  // hoist the loop-invariant B fragments: 2 ksteps x 4 n-tiles = 64 VGPRs
  v16h BW[2][4];
#pragma unroll
  for (int ks = 0; ks < 2; ++ks)
#pragma unroll
    for (int ti = 0; ti < 4; ++ti) BW[ks][ti] = load_b_w(Wt, ti * 16, ks * 32, lane);

  const v8f z8 = {};
  for (int tg = blockIdx.x; tg < MTILES / 8; tg += gridDim.x) {
    int row0 = (tg * 8 + wave) * 16;
#if defined(__gfx1250__)
    if (tg + (int)gridDim.x < MTILES / 8) {  // global_prefetch_b8 of next tile's rows
      const float* pf = X + (size_t)((tg + gridDim.x) * 8 + wave) * 16 * CC;
      __builtin_prefetch(pf, 0, 1);
    }
#endif
    v8f acc[4];
#pragma unroll
    for (int ti = 0; ti < 4; ++ti) acc[ti] = z8;
#pragma unroll
    for (int ks = 0; ks < 2; ++ks) {
      v16h A = load_a_f32(X, row0, ks * 32, lane);
#pragma unroll
      for (int ti = 0; ti < 4; ++ti) acc[ti] = wmma_f16(A, BW[ks][ti], acc[ti]);
    }
    // per-channel partials: all 8 accum slots of a lane share one channel
#pragma unroll
    for (int ti = 0; ti < 4; ++ti) {
      float s = 0, q = 0;
#pragma unroll
      for (int r = 0; r < 8; ++r) { float v = acc[ti][r]; s += v; q += v * v; }
      s += __shfl_xor(s, 16, 32);
      q += __shfl_xor(q, 16, 32);
      if (lane < 16) {
        int c = ti * 16 + lane;
        lds[0][wave][c] += s;
        lds[1][wave][c] += q;
      }
    }
  }
  __syncthreads();
  if (threadIdx.x < 128) {  // 2 quantities x 64 channels
    int q = threadIdx.x >> 6, ch = threadIdx.x & 63;
    float s = 0;
#pragma unroll
    for (int w = 0; w < 8; ++w) s += lds[q][w][ch];
    bp[((size_t)q * 1024 + blockIdx.x) * 64 + ch] = s;
  }
}

// finalize bn0/bn1 stats (deterministic, fixed order). 1 block x 128 threads.
__global__ void k1b_finalize(const float* __restrict__ bp1, float* __restrict__ stats) {
  int t = threadIdx.x; if (t >= 128) return;
  int mat = t >> 6, ch = t & 63;
  float s = 0, q = 0;
  for (int b = 0; b < 1024; ++b) {
    s += bp1[((size_t)(mat * 2 + 0) * 1024 + b) * 64 + ch];
    q += bp1[((size_t)(mat * 2 + 1) * 1024 + b) * 64 + ch];
  }
  float mean = s / (float)MROWS;
  float var  = fmaxf(q / (float)MROWS - mean * mean, 0.0f);
  stats[mat * 128 + ch]      = mean;
  stats[mat * 128 + 64 + ch] = rsqrtf(var + BN_EPS_F);
}

// ---------------------------------------------------------------------------
// K2: WMMA recompute pass + fused BN + ReLU + 64->1 dots (conv2 on relu(x1+x2),
// conv3 on x2), plus partial stats of the two scalar maps.  grid 4096 x 256.
// B fragments are cooperatively packed to f16 in LDS (2 fragments per wave),
// then consumed via 32-byte ds loads — 8x fewer global weight loads per block.
// ---------------------------------------------------------------------------
__global__ void k2_gemm_fused(const float* __restrict__ input, const float* __restrict__ residual,
                              const float* __restrict__ w0, const float* __restrict__ w1,
                              const float* __restrict__ w2v, const float* __restrict__ w3v,
                              const float* __restrict__ stats,
                              const float* __restrict__ sc0, const float* __restrict__ of0,
                              const float* __restrict__ sc1, const float* __restrict__ of1,
                              float* __restrict__ z2raw, float* __restrict__ g3raw,
                              float* __restrict__ bp2) {
  int wave = threadIdx.x >> 5, lane = threadIdx.x & 31;
  int row0 = (blockIdx.x * 8 + wave) * 16;

  // ---- stage the 16 B fragments (2 mats x 2 ksteps x 4 n-tiles) in LDS ----
  __shared__ __align__(32) _Float16 lbf[16][32][16];  // 16 KB
  for (int f = wave; f < 16; f += 8) {                // 2 fragments per wave
    const float* Wm = (f < 8) ? w0 : w1;
    int ks = (f >> 2) & 1, ti = f & 3;
    v16h b = load_b_w(Wm, ti * 16, ks * 32, lane);
    *(v16h*)&lbf[f][lane][0] = b;
  }
  __syncthreads();

  const v8f z8 = {};
  v8f a1c[4], a2c[4];
#pragma unroll
  for (int ti = 0; ti < 4; ++ti) { a1c[ti] = z8; a2c[ti] = z8; }
#pragma unroll
  for (int ks = 0; ks < 2; ++ks) {
    v16h A1 = load_a_f32(input, row0, ks * 32, lane);
    v16h A2 = load_a_f32(residual, row0, ks * 32, lane);
#pragma unroll
    for (int ti = 0; ti < 4; ++ti) {
      v16h B0 = *(const v16h*)&lbf[ks * 4 + ti][lane][0];
      v16h B1 = *(const v16h*)&lbf[8 + ks * 4 + ti][lane][0];
      a1c[ti] = wmma_f16(A1, B0, a1c[ti]);
      a2c[ti] = wmma_f16(A2, B1, a2c[ti]);
    }
  }

  float zp[8], gp[8];
#pragma unroll
  for (int r = 0; r < 8; ++r) { zp[r] = 0.0f; gp[r] = 0.0f; }
#pragma unroll
  for (int ti = 0; ti < 4; ++ti) {
    int c = ti * 16 + (lane & 15);
    float mu0 = stats[S_MEAN0 + c], rs0 = stats[S_RSTD0 + c];
    float mu1 = stats[S_MEAN1 + c], rs1 = stats[S_RSTD1 + c];
    float s0 = sc0[c], o0 = of0[c], s1v = sc1[c], o1 = of1[c];
    float wc2 = w2v[c], wc3 = w3v[c];
#pragma unroll
    for (int r = 0; r < 8; ++r) {
      float x1 = (a1c[ti][r] - mu0) * rs0 * s0 + o0;
      float x2 = (a2c[ti][r] - mu1) * rs1 * s1v + o1;
      float xr = fmaxf(x1 + x2, 0.0f);
      zp[r] += xr * wc2;   // conv2 dot (bias cancels in bn2)
      gp[r] += x2 * wc3;   // conv3 dot on x2 (bias cancels in bn3)
    }
  }
  // reduce over the 16 lanes of each half-group (lanes 0-15 <-> rows r, 16-31 <-> rows r+8)
#pragma unroll
  for (int m = 1; m < 16; m <<= 1)
#pragma unroll
    for (int r = 0; r < 8; ++r) {
      zp[r] += __shfl_xor(zp[r], m, 32);
      gp[r] += __shfl_xor(gp[r], m, 32);
    }

  __shared__ float l2[4][16];
  if (lane == 0 || lane == 16) {
    int half = lane >> 4;
    int base = row0 + half * 8;
    float sz = 0, qz = 0, sg = 0, qg = 0;
#pragma unroll
    for (int r = 0; r < 8; ++r) {
      z2raw[base + r] = zp[r]; g3raw[base + r] = gp[r];
      sz += zp[r]; qz += zp[r] * zp[r]; sg += gp[r]; qg += gp[r] * gp[r];
    }
    int slot = wave * 2 + half;
    l2[0][slot] = sz; l2[1][slot] = qz; l2[2][slot] = sg; l2[3][slot] = qg;
  }
  __syncthreads();
  if (threadIdx.x == 0) {
#pragma unroll
    for (int q = 0; q < 4; ++q) {
      float s = 0;
#pragma unroll
      for (int k = 0; k < 16; ++k) s += l2[q][k];
      bp2[(size_t)q * 4096 + blockIdx.x] = s;
    }
  }
}

// finalize bn2 / bn3 stats. 1 block x 2 threads.
__global__ void k3_finalize(const float* __restrict__ bp2, float* __restrict__ stats) {
  int t = threadIdx.x; if (t >= 2) return;
  float s = 0, q = 0;
  for (int b = 0; b < 4096; ++b) {
    s += bp2[(size_t)(t * 2 + 0) * 4096 + b];
    q += bp2[(size_t)(t * 2 + 1) * 4096 + b];
  }
  float mean = s / (float)MROWS;
  float var  = fmaxf(q / (float)MROWS - mean * mean, 0.0f);
  stats[S_M2 + 2 * t] = mean;
  stats[S_R2 + 2 * t] = rsqrtf(var + BN_EPS_F);
}

// ---------------------------------------------------------------------------
// Bilinear 2x downsample (exact for scale 0.5: average of 2x2) + sigmoid.
// ---------------------------------------------------------------------------
__global__ void k_resize_sigmoid(const float* __restrict__ img, float* __restrict__ imgr) {
  int idx = blockIdx.x * 256 + threadIdx.x;  // < NPIX
  int b = idx >> 16, i = (idx >> 8) & 255, j = idx & 255;
  const float* p = img + (size_t)b * 512 * 512;
  int i2 = i * 2, j2 = j * 2;
  float v = 0.25f * (p[i2 * 512 + j2] + p[i2 * 512 + j2 + 1] +
                     p[(i2 + 1) * 512 + j2] + p[(i2 + 1) * 512 + j2 + 1]);
  imgr[idx] = sigmoidf_(v);
}

// ---------------------------------------------------------------------------
// K4: x = sigmoid(bn2(z2raw)); phi0 = x-0.5 -> phiA & edt.
//     t = (img_r + bn3(g3raw)) * w4  (conv4 bias cancels in bn4); partial stats of t.
// ---------------------------------------------------------------------------
__global__ void k4_elementwise(const float* __restrict__ z2raw, const float* __restrict__ g3raw,
                               const float* __restrict__ imgr, const float* __restrict__ stats,
                               const float* __restrict__ sc2, const float* __restrict__ of2,
                               const float* __restrict__ sc3, const float* __restrict__ of3,
                               const float* __restrict__ w4p,
                               float* __restrict__ phiA, float* __restrict__ edt,
                               float* __restrict__ tmap, float* __restrict__ bpT) {
  int idx = blockIdx.x * 256 + threadIdx.x;
  int tid = threadIdx.x;
  float z  = (z2raw[idx] - stats[S_M2]) * stats[S_R2] * sc2[0] + of2[0];
  float ph = sigmoidf_(z) - 0.5f;
  phiA[idx] = ph; edt[idx] = ph;
  float g  = (g3raw[idx] - stats[S_M3]) * stats[S_R3] * sc3[0] + of3[0];
  float tt = (imgr[idx] + g) * w4p[0];
  tmap[idx] = tt;
  __shared__ float aS[256], aQ[256];
  aS[tid] = tt; aQ[tid] = tt * tt; __syncthreads();
  for (int s = 128; s > 0; s >>= 1) {
    if (tid < s) { aS[tid] += aS[tid + s]; aQ[tid] += aQ[tid + s]; }
    __syncthreads();
  }
  if (tid == 0) { bpT[blockIdx.x] = aS[0]; bpT[2048 + blockIdx.x] = aQ[0]; }
}

__global__ void k5_finalize(const float* __restrict__ bpT, float* __restrict__ stats) {
  if (threadIdx.x != 0) return;
  float s = 0, q = 0;
  for (int b = 0; b < 2048; ++b) { s += bpT[b]; q += bpT[2048 + b]; }
  float mean = s / (float)MROWS;
  float var  = fmaxf(q / (float)MROWS - mean * mean, 0.0f);
  stats[S_M4] = mean; stats[S_R4] = rsqrtf(var + BN_EPS_F);
}

__global__ void k6_gate(const float* __restrict__ tmap, const float* __restrict__ stats,
                        const float* __restrict__ sc4, const float* __restrict__ of4,
                        float* __restrict__ gate) {
  int idx = blockIdx.x * 256 + threadIdx.x;
  gate[idx] = sigmoidf_((tmap[idx] - stats[S_M4]) * stats[S_R4] * sc4[0] + of4[0]);
}

// ---------------------------------------------------------------------------
// conv_phi: one 3x3 zero-padded blur step, accumulated into edt.
// ---------------------------------------------------------------------------
__global__ void k_blur_acc(const float* __restrict__ pin, float* __restrict__ pout,
                           float* __restrict__ edt) {
  int idx = blockIdx.x * 256 + threadIdx.x;
  int b = idx >> 16, i = (idx >> 8) & 255, j = idx & 255;
  const float* p = pin + ((size_t)b << 16);
  float s = 0.0f;
#pragma unroll
  for (int di = -1; di <= 1; ++di)
#pragma unroll
    for (int dj = -1; dj <= 1; ++dj) {
      int ii = i + di, jj = j + dj;
      if (ii >= 0 && ii < 256 && jj >= 0 && jj < 256) s += p[(ii << 8) + jj];
    }
  float v = s * (1.0f / 9.0f);
  pout[idx] = v;
  edt[idx] += v;
}

// ---------------------------------------------------------------------------
// Chan-Vese: per-batch reduction (c1,c2), then stencil update with circular wrap.
// ---------------------------------------------------------------------------
__global__ void k_cv_reduce(const float* __restrict__ phi, const float* __restrict__ gate,
                            float* __restrict__ c1c2) {
  int b = blockIdx.x, tid = threadIdx.x;
  const float* p = phi + ((size_t)b << 16);
  const float* g = gate + ((size_t)b << 16);
  float sH = 0, sGH = 0, sG = 0;
  for (int i = tid; i < 65536; i += 256) {
    float hv = 0.5f * (1.0f + (2.0f / PI_F) * atanf(p[i]));
    float gv = g[i];
    sH += hv; sGH += gv * hv; sG += gv;
  }
  __shared__ float aH[256], aGH[256], aG[256];
  aH[tid] = sH; aGH[tid] = sGH; aG[tid] = sG; __syncthreads();
  for (int s = 128; s > 0; s >>= 1) {
    if (tid < s) { aH[tid] += aH[tid + s]; aGH[tid] += aGH[tid + s]; aG[tid] += aG[tid + s]; }
    __syncthreads();
  }
  if (tid == 0) {
    float areaIn = aH[0];
    c1c2[b * 2 + 0] = aGH[0] / (areaIn + EPS_F);
    c1c2[b * 2 + 1] = (aG[0] - aGH[0]) / ((65536.0f - areaIn) + EPS_F);
  }
}

__device__ __forceinline__ float2 n_at(const float* p, int i, int j) {
  float px = 0.5f * (p[(i << 8) + wrap256(j + 1)] - p[(i << 8) + wrap256(j - 1)]);
  float py = 0.5f * (p[(wrap256(i + 1) << 8) + j] - p[(wrap256(i - 1) << 8) + j]);
  float inv = rsqrtf(px * px + py * py + EPS_F);
  return make_float2(px * inv, py * inv);
}

__global__ void k_cv_update(const float* __restrict__ phiIn, const float* __restrict__ gate,
                            const float* __restrict__ c1c2, float* __restrict__ phiOut) {
  int idx = blockIdx.x * 256 + threadIdx.x;
  int b = idx >> 16, i = (idx >> 8) & 255, j = idx & 255;
  const float* p = phiIn + ((size_t)b << 16);
  float c1 = c1c2[b * 2 + 0], c2 = c1c2[b * 2 + 1];
  float2 nR = n_at(p, i, wrap256(j + 1));
  float2 nL = n_at(p, i, wrap256(j - 1));
  float2 nD = n_at(p, wrap256(i + 1), j);
  float2 nU = n_at(p, wrap256(i - 1), j);
  float curv = 0.5f * (nR.x - nL.x) + 0.5f * (nD.y - nU.y);
  float phic = p[(i << 8) + j];
  float delta = 1.0f / (PI_F * (1.0f + phic * phic));
  float gv = gate[idx];
  float d1 = gv - c1, d2 = gv - c2;
  float F = delta * (0.1f * curv - d1 * d1 + d2 * d2);  // mu=0.1, l1=l2=1
  phiOut[idx] = phic + 0.1f * F;                         // dt=0.1
}

// seg = clip(phi,-1,1); out0 = seg * residual (broadcast over 64 channels)
__global__ void k_final(const float* __restrict__ phi, const float* __restrict__ residual,
                        float* __restrict__ out0, float* __restrict__ seg) {
  size_t t = (size_t)blockIdx.x * 256 + threadIdx.x;  // < NPIX*64
  size_t m = t >> 6;
  float s = fminf(fmaxf(phi[m], -1.0f), 1.0f);
  out0[t] = s * residual[t];
  if ((t & 63) == 0) seg[m] = s;
}

// ---------------------------------------------------------------------------
// Launch. Input order: input, residual, img, is_training, then params pytree
// flattened with dict keys sorted (JAX convention):
// bn0{offset,scale} bn1{..} bn2{..} bn3{..} bn4{..}
// conv0{b,w} conv1{b,w} conv2{b,w} conv3{b,w} conv4{b,w}
// ---------------------------------------------------------------------------
extern "C" void kernel_launch(void* const* d_in, const int* in_sizes, int n_in,
                              void* d_out, int out_size, void* d_ws, size_t ws_size,
                              hipStream_t stream) {
  (void)in_sizes; (void)n_in; (void)out_size; (void)ws_size;
  const float* input    = (const float*)d_in[0];
  const float* residual = (const float*)d_in[1];
  const float* img      = (const float*)d_in[2];
  // d_in[3] = is_training (unused; reference ignores it)
  const float* bn0_of = (const float*)d_in[4];
  const float* bn0_sc = (const float*)d_in[5];
  const float* bn1_of = (const float*)d_in[6];
  const float* bn1_sc = (const float*)d_in[7];
  const float* bn2_of = (const float*)d_in[8];
  const float* bn2_sc = (const float*)d_in[9];
  const float* bn3_of = (const float*)d_in[10];
  const float* bn3_sc = (const float*)d_in[11];
  const float* bn4_of = (const float*)d_in[12];
  const float* bn4_sc = (const float*)d_in[13];
  const float* w0 = (const float*)d_in[15];  // conv0.w [64,64]
  const float* w1 = (const float*)d_in[17];  // conv1.w [64,64]
  const float* w2 = (const float*)d_in[19];  // conv2.w [64]
  const float* w3 = (const float*)d_in[21];  // conv3.w [64]
  const float* w4 = (const float*)d_in[23];  // conv4.w [1]

  float* out0 = (float*)d_out;                       // [B,H,W,64]
  float* edt  = out0 + (size_t)NPIX * CC;            // [B,H,W,1]
  float* seg  = edt + NPIX;                          // [B,H,W,1]
  float* gate = seg + NPIX;                          // [B,H,W,1]

  float* ws    = (float*)d_ws;
  float* phiA  = ws;                 // NPIX
  float* phiB  = phiA + NPIX;        // NPIX
  float* z2raw = phiB + NPIX;        // NPIX
  float* g3raw = z2raw + NPIX;       // NPIX
  float* imgr  = g3raw + NPIX;       // NPIX
  float* tmap  = imgr + NPIX;        // NPIX
  float* bp1   = tmap + NPIX;        // 4*1024*64
  float* bp2   = bp1 + 4 * 1024 * 64;// 4*4096
  float* bpT   = bp2 + 4 * 4096;     // 2*2048
  float* stats = bpT + 2 * 2048;     // 320
  float* c1c2  = stats + 320;        // 16

  // Pass 1: WMMA GEMMs -> bn0/bn1 statistics (one matrix per launch, no spill).
  k1_gemm_stats<<<1024, 256, 0, stream>>>(input, w0, bp1);
  k1_gemm_stats<<<1024, 256, 0, stream>>>(residual, w1, bp1 + (size_t)2 * 1024 * 64);
  k1b_finalize<<<1, 128, 0, stream>>>(bp1, stats);
  // Pass 2: WMMA recompute + fused BN/ReLU/dot -> z2raw, g3raw + their stats.
  k2_gemm_fused<<<4096, 256, 0, stream>>>(input, residual, w0, w1, w2, w3, stats,
                                          bn0_sc, bn0_of, bn1_sc, bn1_of,
                                          z2raw, g3raw, bp2);
  k3_finalize<<<1, 2, 0, stream>>>(bp2, stats);
  k_resize_sigmoid<<<NPIX / 256, 256, 0, stream>>>(img, imgr);
  k4_elementwise<<<NPIX / 256, 256, 0, stream>>>(z2raw, g3raw, imgr, stats,
                                                 bn2_sc, bn2_of, bn3_sc, bn3_of, w4,
                                                 phiA, edt, tmap, bpT);
  k5_finalize<<<1, 32, 0, stream>>>(bpT, stats);
  k6_gate<<<NPIX / 256, 256, 0, stream>>>(tmap, stats, bn4_sc, bn4_of, gate);

  // conv_phi: 5 iterated 3x3 blurs accumulated into edt (phi0 already in phiA/edt).
  float* pa = phiA; float* pb = phiB;
  for (int it = 0; it < 5; ++it) {
    k_blur_acc<<<NPIX / 256, 256, 0, stream>>>(pa, pb, edt);
    float* t = pa; pa = pb; pb = t;
  }

  // Chan-Vese: phi0 = edt; 30 iterations of (reduction, update).
  hipMemcpyAsync(pa, edt, (size_t)NPIX * sizeof(float), hipMemcpyDeviceToDevice, stream);
  for (int it = 0; it < 30; ++it) {
    k_cv_reduce<<<BQ, 256, 0, stream>>>(pa, gate, c1c2);
    k_cv_update<<<NPIX / 256, 256, 0, stream>>>(pa, gate, c1c2, pb);
    float* t = pa; pa = pb; pb = t;
  }

  // seg = clip(phi,-1,1); out0 = seg * residual.
  k_final<<<(NPIX * CC) / 256, 256, 0, stream>>>(pa, residual, out0, seg);
}
